// EfficientLPSSemanticHead_72541997629593
// MI455X (gfx1250) — matrix operand order repaired and checked by reading, not compiled, not run on hardware
//
#include <hip/hip_runtime.h>
#include <hip/hip_bf16.h>

// ---------------------------------------------------------------------------
// CDNA5 (gfx1250) types for WMMA
// ---------------------------------------------------------------------------
typedef __attribute__((ext_vector_type(16))) __bf16          v16bf;
typedef __attribute__((ext_vector_type(16))) unsigned short  v16us;
typedef __attribute__((ext_vector_type(8)))  unsigned short  v8us;
typedef __attribute__((ext_vector_type(4)))  unsigned int    v4u;
typedef __attribute__((ext_vector_type(8)))  float           v8f;

__device__ __forceinline__ unsigned short f32_to_bf16(float f) {
  unsigned int u = __builtin_bit_cast(unsigned int, f);
  unsigned int r = u + 0x7FFFu + ((u >> 16) & 1u);   // round-to-nearest-even
  return (unsigned short)(r >> 16);
}

// ---------------------------------------------------------------------------
// One-time weight conversion fp32 -> bf16, rows padded to M-tile multiple
// (pad rows are zero so the GEMM A-path needs no divergence).
// ---------------------------------------------------------------------------
__global__ void k_cvt_w(const float* __restrict__ x, unsigned short* __restrict__ y,
                        long nvalid, long ntot)
{
  long i = (long)blockIdx.x * blockDim.x + threadIdx.x;
  if (i >= ntot) return;
  y[i] = (i < nvalid) ? f32_to_bf16(x[i]) : (unsigned short)0;
}

// ---------------------------------------------------------------------------
// Transpose-convert activations: [Cin][N] fp32 -> [N][Cin] bf16, per batch.
// LDS-tiled so both global reads and writes are coalesced.
// ---------------------------------------------------------------------------
__global__ __launch_bounds__(256)
void k_tcvt(const float* __restrict__ x, long xbs,
            unsigned short* __restrict__ xt, long tbs, int C, int N)
{
  __shared__ unsigned short tile[32][33];
  int b  = blockIdx.z;
  int n0 = blockIdx.x * 32;
  int c0 = blockIdx.y * 32;
  const float* xb = x + (long)b * xbs;
  for (int r = threadIdx.y; r < 32; r += 8) {
    int c = c0 + r, n = n0 + threadIdx.x;
    tile[r][threadIdx.x] = (c < C && n < N) ? f32_to_bf16(xb[(long)c * N + n])
                                            : (unsigned short)0;
  }
  __syncthreads();
  unsigned short* tb = xt + (long)b * tbs;
  for (int r = threadIdx.y; r < 32; r += 8) {
    int n = n0 + r, c = c0 + threadIdx.x;
    if (n < N && c < C) tb[(long)n * C + c] = tile[threadIdx.x][r];
  }
}

// ---------------------------------------------------------------------------
// Pointwise (1x1) conv as bf16 WMMA GEMM:  Y[co,n] = sum_ci W[co,ci]*X[ci,n]+b
// One wave per 16x16 output tile, K stepped by 32.  Both operands are K-major
// bf16, so each fragment is two contiguous 16B chunks -> 4x global_load_b128
// + 1x v_wmma_f32_16x16x32_bf16 per K step, no conversion VALU in the loop.
// Per-lane layouts per CDNA5 ISA §7.12.2 (wave32):
//   A 16x32 bf16 : M = lane%16 ; e<8 -> K = kb+e, e>=8 -> K = kb+16+(e-8),
//                  kb = (lane<16) ? 0 : 8
//   B 32x16 bf16 : N = lane%16 ; same K order as A (e<8 -> kb+e, e>=8 -> kb+16+(e-8))
//   D 16x16 f32  : elem r -> M = m0 + (lane<16?0:8) + r, N = lane%16
// ---------------------------------------------------------------------------
__global__ __launch_bounds__(32)
void k_pw_gemm(const unsigned short* __restrict__ Xt, long xbs,  // [N][Cin] bf16
               const unsigned short* __restrict__ Wb,            // [CoutPad][Cin] bf16
               const float* __restrict__ bias,
               float* __restrict__ Y, long ybs,
               int Cin, int Cout, int N)
{
  const int lane = threadIdx.x;
  const int n0   = blockIdx.x * 16;
  const int m0   = blockIdx.y * 16;
  const int b    = blockIdx.z;

  const unsigned short* Xtb = Xt + (long)b * xbs;
  float*                Yb  = Y + (long)b * ybs;

  const int half = lane >> 4;      // 0 or 1
  const int l16  = lane & 15;
  const int m    = m0 + l16;       // A row owned by this lane (padded rows valid)
  const int n    = n0 + l16;       // B / D column owned by this lane
  const int kb   = half * 8;

  const unsigned short* wr = Wb + (long)m * Cin + kb;
  const unsigned short* xr = Xtb + (long)n * Cin + kb;

  v8f acc;
#pragma unroll
  for (int r = 0; r < 8; ++r) acc[r] = 0.0f;

  for (int k0 = 0; k0 < Cin; k0 += 32) {
    v4u a0 = *(const v4u*)(wr + k0);        // A elems 0..7   (K = kb..kb+7)
    v4u a1 = *(const v4u*)(wr + k0 + 16);   // A elems 8..15  (K = kb+16..kb+23)
    v4u b0 = *(const v4u*)(xr + k0);        // B elems 0..7   (K = kb..kb+7)
    v4u b1 = *(const v4u*)(xr + k0 + 16);   // B elems 8..15  (K = kb+16..kb+23)

    if (k0 + 32 < Cin)  // gfx1250 global_prefetch of next K panel
      __builtin_prefetch(xr + k0 + 32, 0, 1);

    v8us a0s = __builtin_bit_cast(v8us, a0);
    v8us a1s = __builtin_bit_cast(v8us, a1);
    v8us b0s = __builtin_bit_cast(v8us, b0);
    v8us b1s = __builtin_bit_cast(v8us, b1);
    v16us au = __builtin_shufflevector(a0s, a1s, 0,1,2,3,4,5,6,7,8,9,10,11,12,13,14,15);
    v16us bu = __builtin_shufflevector(b0s, b1s, 0,1,2,3,4,5,6,7,8,9,10,11,12,13,14,15);
    v16bf av = __builtin_bit_cast(v16bf, au);
    v16bf bv = __builtin_bit_cast(v16bf, bu);
    acc = __builtin_amdgcn_wmma_f32_16x16x32_bf16(
        /*neg_a=*/false, av, /*neg_b=*/false, bv,
        /*c_mod=*/(short)0, acc, /*reuse_a=*/false, /*reuse_b=*/false);
  }

#pragma unroll
  for (int r = 0; r < 8; ++r) {
    int mm = m0 + half * 8 + r;
    if (mm < Cout) Yb[(long)mm * N + n] = acc[r] + bias[mm];
  }
}

// ---------------------------------------------------------------------------
// Depthwise 3x3 conv, dilation == padding (spatial size preserved)
// ---------------------------------------------------------------------------
__global__ void k_dw3x3(const float* __restrict__ x, long xbs,
                        const float* __restrict__ wt, const float* __restrict__ bs,
                        float* __restrict__ y, long ybs,
                        int C, int H, int W, int dh, int dwl)
{
  long i = (long)blockIdx.x * blockDim.x + threadIdx.x;
  long total = 2L * C * H * W;
  if (i >= total) return;
  int w = (int)(i % W); long r = i / W;
  int h = (int)(r % H); r /= H;
  int c = (int)(r % C); int b = (int)(r / C);

  const float* xc = x + (long)b * xbs + (long)c * H * W;
  const float* wc = wt + (long)c * 9;
  float acc = bs[c];
#pragma unroll
  for (int kh = 0; kh < 3; ++kh) {
    int hh = h + (kh - 1) * dh;
    if (hh < 0 || hh >= H) continue;
#pragma unroll
    for (int kw = 0; kw < 3; ++kw) {
      int ww = w + (kw - 1) * dwl;
      if (ww < 0 || ww >= W) continue;
      acc += xc[(long)hh * W + ww] * wc[kh * 3 + kw];
    }
  }
  y[(long)b * ybs + ((long)c * H + h) * W + w] = acc;
}

// ---------------------------------------------------------------------------
// Full 3x3 conv, pad 1, dilation 1 (tiny Cout=2 range-offset convs)
// ---------------------------------------------------------------------------
__global__ void k_conv3x3(const float* __restrict__ x, long xbs,
                          const float* __restrict__ wt, const float* __restrict__ bs,
                          float* __restrict__ y, long ybs,
                          int Cin, int Cout, int H, int W)
{
  long i = (long)blockIdx.x * blockDim.x + threadIdx.x;
  long total = 2L * Cout * H * W;
  if (i >= total) return;
  int w = (int)(i % W); long r = i / W;
  int h = (int)(r % H); r /= H;
  int co = (int)(r % Cout); int b = (int)(r / Cout);

  float acc = bs[co];
  const float* xb = x + (long)b * xbs;
  const float* wc = wt + (long)co * Cin * 9;
  for (int ci = 0; ci < Cin; ++ci) {
    const float* xc = xb + (long)ci * H * W;
    const float* wk = wc + ci * 9;
#pragma unroll
    for (int kh = 0; kh < 3; ++kh) {
      int hh = h + kh - 1;
      if (hh < 0 || hh >= H) continue;
#pragma unroll
      for (int kw = 0; kw < 3; ++kw) {
        int ww = w + kw - 1;
        if (ww < 0 || ww >= W) continue;
        acc += xc[(long)hh * W + ww] * wk[kh * 3 + kw];
      }
    }
  }
  y[(long)b * ybs + ((long)co * H + h) * W + w] = acc;
}

// ---------------------------------------------------------------------------
// Global min/max of the two channels of x_range (B,2,H,W) -> out4={mn0,mx0,mn1,mx1}
// ---------------------------------------------------------------------------
__global__ void k_minmax2(const float* __restrict__ xr, int HW, int B,
                          float* __restrict__ out4)
{
  __shared__ float s0[256], s1[256], s2[256], s3[256];
  int tid = threadIdx.x;
  float mn0 = 3.4e38f, mx0 = -3.4e38f, mn1 = 3.4e38f, mx1 = -3.4e38f;
  for (int i = tid; i < B * HW; i += 256) {
    int b = i / HW, j = i % HW;
    float a = xr[((long)b * 2 + 0) * HW + j];
    float c = xr[((long)b * 2 + 1) * HW + j];
    mn0 = fminf(mn0, a); mx0 = fmaxf(mx0, a);
    mn1 = fminf(mn1, c); mx1 = fmaxf(mx1, c);
  }
  s0[tid] = mn0; s1[tid] = mx0; s2[tid] = mn1; s3[tid] = mx1;
  __syncthreads();
  for (int s = 128; s > 0; s >>= 1) {
    if (tid < s) {
      s0[tid] = fminf(s0[tid], s0[tid + s]);
      s1[tid] = fmaxf(s1[tid], s1[tid + s]);
      s2[tid] = fminf(s2[tid], s2[tid + s]);
      s3[tid] = fmaxf(s3[tid], s3[tid + s]);
    }
    __syncthreads();
  }
  if (tid == 0) { out4[0] = s0[0]; out4[1] = s1[0]; out4[2] = s2[0]; out4[3] = s3[0]; }
}

// ---------------------------------------------------------------------------
// Zero-pad (B,C,H,W) -> compact (B,C,H+2,W+2)
// ---------------------------------------------------------------------------
__global__ void k_pad(const float* __restrict__ x, long xbs,
                      float* __restrict__ xp, int C, int H, int W)
{
  int Hp = H + 2, Wp = W + 2;
  long i = (long)blockIdx.x * blockDim.x + threadIdx.x;
  long total = 2L * C * Hp * Wp;
  if (i >= total) return;
  int wp = (int)(i % Wp); long r = i / Wp;
  int hp = (int)(r % Hp); r /= Hp;
  int c = (int)(r % C); int b = (int)(r / C);
  float v = 0.0f;
  if (hp >= 1 && hp <= H && wp >= 1 && wp <= W)
    v = x[(long)b * xbs + ((long)c * H + (hp - 1)) * W + (wp - 1)];
  xp[(((long)b * C + c) * Hp + hp) * Wp + wp] = v;
}

// ---------------------------------------------------------------------------
// shift_x stage 1: per (b, pixel, tap) gather indices + combined weights
// (channel-invariant; reused across all C channels)
// ---------------------------------------------------------------------------
__global__ void k_shift_idx(const float* __restrict__ xr, const float* __restrict__ mm,
                            int H, int W, float dmax,
                            int4* __restrict__ idx4, float4* __restrict__ w4)
{
  long i = (long)blockIdx.x * blockDim.x + threadIdx.x;
  int HW = H * W;
  long total = 2L * HW * 9;
  if (i >= total) return;
  int t = (int)(i % 9); long r = i / 9;
  int hw = (int)(r % HW); int b = (int)(r / HW);
  int Wp = W + 2, Hp = H + 2;

  float xo = (float)(t % 3 - 1);
  float yo = (float)(t / 3 - 1);
  float mn0 = mm[0], mx0 = mm[1], mn1 = mm[2], mx1 = mm[3];
  float ox = dmax * (xr[((long)b * 2 + 0) * HW + hw] - mn0) / (mx0 - mn0);
  float oy = dmax * (xr[((long)b * 2 + 1) * HW + hw] - mn1) / (mx1 - mn1);

  int h = hw / W, w = hw % W;
  float pre   = (float)((h + 1) * Wp + (w + 1)) + xo + yo * (float)W + 2.0f;
  float off   = ox * xo + oy * (yo * (float)Wp);
  float after = pre + off;
  float mxv   = (float)(Hp * Wp - 1);

  float a_f  = fminf(fmaxf(pre + floorf(off), 0.f), mxv);
  float a_f1 = fminf(fmaxf(a_f + xo,          0.f), mxv);
  float a_c  = fminf(fmaxf(pre + ceilf(off),  0.f), mxv);
  float a_c1 = fminf(fmaxf(a_c + xo,          0.f), mxv);

  float wA = fabsf((after - a_f)  / (float)Wp);
  float wB = fabsf((a_c1 - after) / (float)Wp);

  idx4[i] = make_int4((int)a_f, (int)a_f1, (int)a_c1, (int)a_c);
  w4[i]   = make_float4(wA * fabsf(after - a_f), wA * fabsf(a_f1 - after),
                        wB * fabsf(after - a_c1), wB * fabsf(a_c - after));
}

// ---------------------------------------------------------------------------
// shift_x stage 2: apply gathers + range_out(C,9) reduction
// ---------------------------------------------------------------------------
__global__ void k_shift_apply(const float* __restrict__ pad,
                              const int4* __restrict__ idx4,
                              const float4* __restrict__ w4,
                              const float* __restrict__ ro,
                              float* __restrict__ out,
                              int C, int HW, int HpWp)
{
  long i = (long)blockIdx.x * blockDim.x + threadIdx.x;
  long total = 2L * C * HW;
  if (i >= total) return;
  int hw = (int)(i % HW); long r = i / HW;
  int c = (int)(r % C); int b = (int)(r / C);
  const float* xp = pad + ((long)b * C + c) * HpWp;
  long base = ((long)b * HW + hw) * 9;
  float acc = 0.0f;
#pragma unroll
  for (int t = 0; t < 9; ++t) {
    int4   id = idx4[base + t];
    float4 wv = w4[base + t];
    acc += ro[c * 9 + t] * (wv.x * xp[id.x] + wv.y * xp[id.y] +
                            wv.z * xp[id.z] + wv.w * xp[id.w]);
  }
  out[((long)b * C + c) * HW + hw] = acc;
}

// ---------------------------------------------------------------------------
// Bilinear resize (half-pixel centers, edge clamp) with sliced dst
// ---------------------------------------------------------------------------
__global__ void k_resize(const float* __restrict__ src, long sbs,
                         float* __restrict__ dst, long dbs,
                         int C, int Hs, int Ws, int Hd, int Wd)
{
  long i = (long)blockIdx.x * blockDim.x + threadIdx.x;
  long total = 2L * C * Hd * Wd;
  if (i >= total) return;
  int wd = (int)(i % Wd); long r = i / Wd;
  int hd = (int)(r % Hd); r /= Hd;
  int c = (int)(r % C); int b = (int)(r / C);

  float sy = (float)Hs / (float)Hd, sx = (float)Ws / (float)Wd;
  float fy = fminf(fmaxf((hd + 0.5f) * sy - 0.5f, 0.f), (float)(Hs - 1));
  float fx = fminf(fmaxf((wd + 0.5f) * sx - 0.5f, 0.f), (float)(Ws - 1));
  int y0 = (int)floorf(fy), x0 = (int)floorf(fx);
  int y1 = min(y0 + 1, Hs - 1), x1 = min(x0 + 1, Ws - 1);
  float ty = fy - y0, tx = fx - x0;

  const float* sc = src + (long)b * sbs + (long)c * Hs * Ws;
  float v = (1.f - ty) * ((1.f - tx) * sc[(long)y0 * Ws + x0] + tx * sc[(long)y0 * Ws + x1]) +
            ty         * ((1.f - tx) * sc[(long)y1 * Ws + x0] + tx * sc[(long)y1 * Ws + x1]);
  dst[(long)b * dbs + ((long)c * Hd + hd) * Wd + wd] = v;
}

// ---------------------------------------------------------------------------
// out[b,:]=a[b,:]+c[b,:] and strided channel-slice copy
// ---------------------------------------------------------------------------
__global__ void k_add2(float* __restrict__ o, long obs,
                       const float* __restrict__ a, long abs_,
                       const float* __restrict__ c, long cbs, long n)
{
  long i = (long)blockIdx.x * blockDim.x + threadIdx.x;
  if (i >= 2 * n) return;
  long b = i / n, j = i % n;
  o[b * obs + j] = a[b * abs_ + j] + c[b * cbs + j];
}

__global__ void k_copy(const float* __restrict__ a, long abs_,
                       float* __restrict__ o, long obs, long n)
{
  long i = (long)blockIdx.x * blockDim.x + threadIdx.x;
  if (i >= 2 * n) return;
  long b = i / n, j = i % n;
  o[b * obs + j] = a[b * abs_ + j];
}

// ---------------------------------------------------------------------------
// Param index map (d_in[6 + idx]), following make_params insertion order
// ---------------------------------------------------------------------------
enum {
  DPC0 = 0,                                // c_j at DPC0 + 4*j : dw_w,dw_b,pw_w,pw_b
  DPC0_OUT_W = 20, DPC0_OUT_B = 21,
  DPC0_ROA = 22, DPC0_ROB = 23, DPC0_RCA_W = 24, DPC0_RCA_B = 25,
  DPC0_MERGE_W = 26, DPC0_MERGE_B = 27,
  DPC1 = 28, DPC1_OUT_W = 48, DPC1_OUT_B = 49,
  LSFE0_RO = 50, LSFE0_RC_W = 51, LSFE0_RC_B = 52,
  LSFE0_C0 = 53, LSFE0_C1 = 57,
  LSFE1_C0 = 61, LSFE1_C1 = 65,
  MC0_C0 = 69, MC0_C1 = 73,
  MC1_C0 = 77, MC1_C1 = 81,
  RO0_W = 85, RO0_B = 86, RO1_W = 87, RO1_B = 88,
  LOGITS_W = 89, LOGITS_B = 90,
  NPARAM = 91
};

extern "C" void kernel_launch(void* const* d_in, const int* in_sizes, int n_in,
                              void* d_out, int out_size, void* d_ws, size_t ws_size,
                              hipStream_t stream)
{
  const float* f0 = (const float*)d_in[0];   // (2,256,16,512)
  const float* f1 = (const float*)d_in[1];   // (2,256,8,256)
  const float* f2 = (const float*)d_in[2];   // (2,256,4,128)
  const float* f3 = (const float*)d_in[3];   // (2,256,2,64)
  const float* r1 = (const float*)d_in[4];   // (2,8,8,256)
  const float* r3 = (const float*)d_in[5];   // (2,128,2,64)
  const float* P[NPARAM];
  for (int i = 0; i < NPARAM; ++i) P[i] = (const float*)d_in[6 + i];

  const int H0 = 16, W0 = 512, HW0 = 8192;
  const int H1 = 8,  W1 = 256, HW1 = 2048;
  const int H2 = 4,  W2 = 128, HW2 = 512;
  const int H3 = 2,  W3 = 64,  HW3 = 128;

  size_t off = 0;
  char* wsb = (char*)d_ws;
  auto allocB = [&](size_t bytes) -> void* {
    void* p = wsb + off;
    off = (off + bytes + 255) & ~(size_t)255;
    return p;
  };
  auto allocF = [&](long elems) -> float* { return (float*)allocB((size_t)elems * 4); };
  auto blks = [](long n) { return dim3((unsigned)((n + 255) / 256)); };

  // ---- one-time bf16 weight conversion (rows padded to multiple of 16) ----
  unsigned short* WQ[NPARAM] = {};
  auto cvtW = [&](int idx, int Cout, int Cin) {
    int cp = (Cout + 15) & ~15;
    unsigned short* p = (unsigned short*)allocB((size_t)cp * Cin * 2);
    long ntot = (long)cp * Cin;
    k_cvt_w<<<blks(ntot), 256, 0, stream>>>(P[idx], p, (long)Cout * Cin, ntot);
    WQ[idx] = p;
  };
  for (int j = 0; j < 5; ++j) cvtW(DPC0 + 4 * j + 2, 256, 256);
  cvtW(DPC0_OUT_W, 128, 1536);
  cvtW(DPC0_RCA_W, 128, 256);
  cvtW(DPC0_MERGE_W, 256, 256);
  for (int j = 0; j < 5; ++j) cvtW(DPC1 + 4 * j + 2, 256, 256);
  cvtW(DPC1_OUT_W, 128, 1280);
  cvtW(LSFE0_RC_W, 128, 256);
  cvtW(LSFE0_C0 + 2, 128, 128);  cvtW(LSFE0_C1 + 2, 128, 128);
  cvtW(LSFE1_C0 + 2, 128, 256);  cvtW(LSFE1_C1 + 2, 128, 128);
  cvtW(MC0_C0 + 2, 128, 128);    cvtW(MC0_C1 + 2, 128, 128);
  cvtW(MC1_C0 + 2, 128, 128);    cvtW(MC1_C1 + 2, 128, 128);
  cvtW(LOGITS_W, 19, 512);

  // ---- shared transposed-bf16 activation scratch (max N*Cin = 8192*512) ----
  unsigned short* xt = (unsigned short*)allocB(2UL * 512 * HW0 * 2);

  auto pw = [&](const float* X, long xbs, int widx, const float* bb,
                float* Y, long ybs, int Cin, int Cout, int N) {
    dim3 tg(N / 32, Cin / 32, 2);
    k_tcvt<<<tg, dim3(32, 8), 0, stream>>>(X, xbs, xt, (long)N * Cin, Cin, N);
    dim3 g(N / 16, (Cout + 15) / 16, 2);
    k_pw_gemm<<<g, 32, 0, stream>>>(xt, (long)N * Cin, WQ[widx], bb, Y, ybs, Cin, Cout, N);
  };
  auto ds_conv = [&](const float* X, long xbs, int pidx, int Cin, int Cout,
                     int H, int W, int dh, int dwl, float* tmp, float* Y, long ybs) {
    long tot = 2L * Cin * H * W;
    k_dw3x3<<<blks(tot), 256, 0, stream>>>(X, xbs, P[pidx], P[pidx + 1],
                                           tmp, (long)Cin * H * W, Cin, H, W, dh, dwl);
    pw(tmp, (long)Cin * H * W, pidx + 2, P[pidx + 3], Y, ybs, Cin, Cout, H * W);
  };

  // ======================= r_off3 = conv3x3(r3, ro1) =======================
  float* roff3 = allocF(2L * 2 * HW3);
  k_conv3x3<<<blks(2L * 2 * HW3), 256, 0, stream>>>(
      r3, (long)128 * HW3, P[RO1_W], P[RO1_B], roff3, (long)2 * HW3, 128, 2, H3, W3);

  // ======================= dpc0 (flag=True) on f3 ==========================
  float* dwt3 = allocF(2L * 256 * HW3);
  float* xu3  = allocF(2L * 256 * HW3);
  float* xf3  = allocF(2L * 1536 * HW3);
  ds_conv(f3, 256L * HW3, DPC0 + 0, 256, 256, H3, W3, 1, 6, dwt3, xu3, 256L * HW3);
  k_copy<<<blks(2L * 256 * HW3), 256, 0, stream>>>(f3, 256L * HW3, xf3, 1536L * HW3, 256L * HW3);
  ds_conv(xu3, 256L * HW3, DPC0 + 4,  256, 256, H3, W3, 1, 1,   dwt3, xf3 + 256L * HW3,  1536L * HW3);
  ds_conv(xu3, 256L * HW3, DPC0 + 8,  256, 256, H3, W3, 6, 21,  dwt3, xf3 + 512L * HW3,  1536L * HW3);
  ds_conv(xu3, 256L * HW3, DPC0 + 12, 256, 256, H3, W3, 18, 15, dwt3, xf3 + 768L * HW3,  1536L * HW3);
  ds_conv(xf3 + 768L * HW3, 1536L * HW3, DPC0 + 16, 256, 256, H3, W3, 6, 3,
          dwt3, xf3 + 1024L * HW3, 1536L * HW3);

  // shift branch
  float* mm3 = allocF(8);
  k_minmax2<<<1, 256, 0, stream>>>(roff3, HW3, 2, mm3);
  int PP3 = (H3 + 2) * (W3 + 2);
  float* padx3  = allocF(2L * 256 * PP3);
  float* padxu3 = allocF(2L * 256 * PP3);
  k_pad<<<blks(2L * 256 * PP3), 256, 0, stream>>>(f3,  256L * HW3, padx3,  256, H3, W3);
  k_pad<<<blks(2L * 256 * PP3), 256, 0, stream>>>(xu3, 256L * HW3, padxu3, 256, H3, W3);
  long taps3 = 2L * HW3 * 9;
  int4*   idx3 = (int4*)allocB((size_t)taps3 * 16);
  float4* wv3  = (float4*)allocB((size_t)taps3 * 16);
  k_shift_idx<<<blks(taps3), 256, 0, stream>>>(roff3, mm3, H3, W3, 24.0f, idx3, wv3);
  float* sx3  = allocF(2L * 256 * HW3);
  float* sxu3 = allocF(2L * 256 * HW3);
  k_shift_apply<<<blks(2L * 256 * HW3), 256, 0, stream>>>(padx3,  idx3, wv3, P[DPC0_ROA], sx3,  256, HW3, PP3);
  k_shift_apply<<<blks(2L * 256 * HW3), 256, 0, stream>>>(padxu3, idx3, wv3, P[DPC0_ROB], sxu3, 256, HW3, PP3);
  float* xab3 = allocF(2L * 256 * HW3);
  pw(sx3,  256L * HW3, DPC0_RCA_W, P[DPC0_RCA_B], xab3,               256L * HW3, 256, 128, HW3);
  pw(sxu3, 256L * HW3, DPC0_RCA_W, P[DPC0_RCA_B], xab3 + 128L * HW3,  256L * HW3, 256, 128, HW3);
  pw(xab3, 256L * HW3, DPC0_MERGE_W, P[DPC0_MERGE_B], xf3 + 1280L * HW3, 1536L * HW3, 256, 256, HW3);

  float* f3o = allocF(2L * 128 * HW3);
  pw(xf3, 1536L * HW3, DPC0_OUT_W, P[DPC0_OUT_B], f3o, 128L * HW3, 1536, 128, HW3);

  // ======================= dpc1 (flag=False) on f2 =========================
  float* dwt2 = allocF(2L * 256 * HW2);
  float* xu2  = allocF(2L * 256 * HW2);
  float* xf2  = allocF(2L * 1280 * HW2);
  ds_conv(f2, 256L * HW2, DPC1 + 0, 256, 256, H2, W2, 1, 6, dwt2, xu2, 256L * HW2);
  k_copy<<<blks(2L * 256 * HW2), 256, 0, stream>>>(f2, 256L * HW2, xf2, 1280L * HW2, 256L * HW2);
  ds_conv(xu2, 256L * HW2, DPC1 + 4,  256, 256, H2, W2, 1, 1,   dwt2, xf2 + 256L * HW2,  1280L * HW2);
  ds_conv(xu2, 256L * HW2, DPC1 + 8,  256, 256, H2, W2, 6, 21,  dwt2, xf2 + 512L * HW2,  1280L * HW2);
  ds_conv(xu2, 256L * HW2, DPC1 + 12, 256, 256, H2, W2, 18, 15, dwt2, xf2 + 768L * HW2,  1280L * HW2);
  ds_conv(xf2 + 768L * HW2, 1280L * HW2, DPC1 + 16, 256, 256, H2, W2, 6, 3,
          dwt2, xf2 + 1024L * HW2, 1280L * HW2);
  float* f2o = allocF(2L * 128 * HW2);
  pw(xf2, 1280L * HW2, DPC1_OUT_W, P[DPC1_OUT_B], f2o, 128L * HW2, 1280, 128, HW2);

  // ======================= mc0( f2o + up(f3o) ) ============================
  float* upt  = allocF(2L * 128 * HW2);
  k_resize<<<blks(2L * 128 * HW2), 256, 0, stream>>>(f3o, 128L * HW3, upt, 128L * HW2, 128, H3, W3, H2, W2);
  float* m0in = allocF(2L * 128 * HW2);
  k_add2<<<blks(2L * 128 * HW2), 256, 0, stream>>>(m0in, 128L * HW2, f2o, 128L * HW2, upt, 128L * HW2, 128L * HW2);
  float* mdw = allocF(2L * 128 * HW2);
  float* ma  = allocF(2L * 128 * HW2);
  float* mb  = allocF(2L * 128 * HW2);
  ds_conv(m0in, 128L * HW2, MC0_C0, 128, 128, H2, W2, 1, 1, mdw, ma, 128L * HW2);
  ds_conv(ma,   128L * HW2, MC0_C1, 128, 128, H2, W2, 1, 1, mdw, mb, 128L * HW2);
  float* mc0out = allocF(2L * 128 * HW1);
  k_resize<<<blks(2L * 128 * HW1), 256, 0, stream>>>(mb, 128L * HW2, mc0out, 128L * HW1, 128, H2, W2, H1, W1);

  // ======================= r_off1 = conv3x3(r1, ro0) =======================
  float* roff1 = allocF(2L * 2 * HW1);
  k_conv3x3<<<blks(2L * 2 * HW1), 256, 0, stream>>>(
      r1, (long)8 * HW1, P[RO0_W], P[RO0_B], roff1, (long)2 * HW1, 8, 2, H1, W1);

  // ======================= lsfe0 (flag=True) on f1 =========================
  float* mm1 = allocF(8);
  k_minmax2<<<1, 256, 0, stream>>>(roff1, HW1, 2, mm1);
  int PP1 = (H1 + 2) * (W1 + 2);
  float* padf1 = allocF(2L * 256 * PP1);
  k_pad<<<blks(2L * 256 * PP1), 256, 0, stream>>>(f1, 256L * HW1, padf1, 256, H1, W1);
  long taps1 = 2L * HW1 * 9;
  int4*   idx1 = (int4*)allocB((size_t)taps1 * 16);
  float4* wv1  = (float4*)allocB((size_t)taps1 * 16);
  k_shift_idx<<<blks(taps1), 256, 0, stream>>>(roff1, mm1, H1, W1, 3.0f, idx1, wv1);
  float* sh1 = allocF(2L * 256 * HW1);
  k_shift_apply<<<blks(2L * 256 * HW1), 256, 0, stream>>>(padf1, idx1, wv1, P[LSFE0_RO], sh1, 256, HW1, PP1);
  float* rc1 = allocF(2L * 128 * HW1);
  pw(sh1, 256L * HW1, LSFE0_RC_W, P[LSFE0_RC_B], rc1, 128L * HW1, 256, 128, HW1);
  float* ldw = allocF(2L * 128 * HW1);
  float* la  = allocF(2L * 128 * HW1);
  float* lb  = allocF(2L * 128 * HW1);
  ds_conv(rc1, 128L * HW1, LSFE0_C0, 128, 128, H1, W1, 1, 1, ldw, la, 128L * HW1);
  ds_conv(la,  128L * HW1, LSFE0_C1, 128, 128, H1, W1, 1, 1, ldw, lb, 128L * HW1);
  // f1 <- lsfe0(f1) + mc0out  (up_to same size == identity)
  float* f1f = allocF(2L * 128 * HW1);
  k_add2<<<blks(2L * 128 * HW1), 256, 0, stream>>>(f1f, 128L * HW1, lb, 128L * HW1, mc0out, 128L * HW1, 128L * HW1);

  // ======================= mc1(f1f) ========================================
  ds_conv(f1f, 128L * HW1, MC1_C0, 128, 128, H1, W1, 1, 1, ldw, la, 128L * HW1);
  ds_conv(la,  128L * HW1, MC1_C1, 128, 128, H1, W1, 1, 1, ldw, lb, 128L * HW1);
  float* mc1out = allocF(2L * 128 * HW0);
  k_resize<<<blks(2L * 128 * HW0), 256, 0, stream>>>(lb, 128L * HW1, mc1out, 128L * HW0, 128, H1, W1, H0, W0);

  // ======== concat buffer (2,512,16,512); slices written in place ==========
  float* cc = allocF(2L * 512 * HW0);

  // lsfe1 (flag=False) on f0 -> concat channels [0,128)
  float* fdw = allocF(2L * 256 * HW0);
  float* g0  = allocF(2L * 128 * HW0);
  ds_conv(f0, 256L * HW0, LSFE1_C0, 256, 128, H0, W0, 1, 1, fdw, g0, 128L * HW0);
  float* gdw = allocF(2L * 128 * HW0);
  ds_conv(g0, 128L * HW0, LSFE1_C1, 128, 128, H0, W0, 1, 1, gdw, cc, 512L * HW0);
  // f0 <- lsfe1(f0) + mc1out (identity resize), accumulated into the slice
  k_add2<<<blks(2L * 128 * HW0), 256, 0, stream>>>(cc, 512L * HW0, cc, 512L * HW0, mc1out, 128L * HW0, 128L * HW0);

  // up_to(f1f) -> [128,256) ; up_to(f2o) -> [256,384) ; up_to(f3o) -> [384,512)
  k_resize<<<blks(2L * 128 * HW0), 256, 0, stream>>>(f1f, 128L * HW1, cc + 128L * HW0, 512L * HW0, 128, H1, W1, H0, W0);
  k_resize<<<blks(2L * 128 * HW0), 256, 0, stream>>>(f2o, 128L * HW2, cc + 256L * HW0, 512L * HW0, 128, H2, W2, H0, W0);
  k_resize<<<blks(2L * 128 * HW0), 256, 0, stream>>>(f3o, 128L * HW3, cc + 384L * HW0, 512L * HW0, 128, H3, W3, H0, W0);

  // ======================= logits: 19x512 WMMA GEMM ========================
  pw(cc, 512L * HW0, LOGITS_W, P[LOGITS_B], (float*)d_out, 19L * HW0, 512, 19, HW0);

  (void)in_sizes; (void)n_in; (void)out_size; (void)ws_size;
}
